// Attention_86431921864795
// MI455X (gfx1250) — matrix operand order repaired and checked by reading, not compiled, not run on hardware
//
#include <hip/hip_runtime.h>
#include <stdint.h>

#define DEV __device__ __forceinline__

typedef __attribute__((ext_vector_type(16))) __bf16 v16bf;
typedef __attribute__((ext_vector_type(8)))  float  v8f;
typedef __attribute__((ext_vector_type(4)))  unsigned int u32x4;
typedef __attribute__((ext_vector_type(8)))  int i32x8;
typedef __attribute__((ext_vector_type(4)))  int i32x4;

union FragBF { u32x4 q[2]; v16bf v; };

DEV unsigned short f2bf(float f) {
  unsigned int u = __float_as_uint(f);
  u += 0x7FFFu + ((u >> 16) & 1u);   // round-to-nearest-even
  return (unsigned short)(u >> 16);
}

DEV v8f vzero8() { v8f v; for (int i = 0; i < 8; ++i) v[i] = 0.f; return v; }

DEV v8f wmma_bf16(v16bf a, v16bf b, v8f c) {
#if defined(__gfx1250__)
  return __builtin_amdgcn_wmma_f32_16x16x32_bf16(false, a, false, b, (short)0, c,
                                                 false, false);
#else
  return c;
#endif
}

DEV float rmax16(float v) {
  v = fmaxf(v, __shfl_xor(v, 1, 32));
  v = fmaxf(v, __shfl_xor(v, 2, 32));
  v = fmaxf(v, __shfl_xor(v, 4, 32));
  v = fmaxf(v, __shfl_xor(v, 8, 32));
  return v;
}
DEV float rsum16(float v) {
  v += __shfl_xor(v, 1, 32);
  v += __shfl_xor(v, 2, 32);
  v += __shfl_xor(v, 4, 32);
  v += __shfl_xor(v, 8, 32);
  return v;
}

#if defined(__gfx1250__) && __has_builtin(__builtin_amdgcn_tensor_load_to_lds)
#define USE_TDM 1
#else
#define USE_TDM 0
#endif

#if USE_TDM
DEV unsigned lds_byte_offset(const void* p) {
  return (unsigned)(unsigned long long)
      (__attribute__((address_space(3))) const void*)p;
}

// TDM: load a 128x64 tile of bf16 from a row-major global tensor into LDS.
// Pad feature inserts 16B after every 64B of stored data, so each 128B data
// row becomes a 160B LDS row -> row stride 80 bf16, sub-block k=32..63 starts
// at element 40. Fragment reads stay 16B-aligned and bank-conflict-free.
DEV void tdm_load_tile_128x64(const unsigned short* gtile, unsigned lds_off,
                              unsigned row_stride_elems) {
  unsigned long long ga = (unsigned long long)(size_t)gtile;
  u32x4 g0;
  g0[0] = 1u;                                        // count=1 valid user D#
  g0[1] = lds_off;                                   // lds_addr (bytes)
  g0[2] = (unsigned)(ga & 0xFFFFFFFFull);            // global_addr[31:0]
  g0[3] = (unsigned)((ga >> 32) & 0x01FFFFFFull)     // global_addr[56:32]
        | 0x80000000u;                               // type=2 ("image")
  i32x8 g1;
  g1[0] = (int)((1u << 16)     // data_size = 2 bytes
              | (1u << 20)     // pad_enable
              | (3u << 22)     // pad_interval: every 16 DWORDs (64B)
              | (3u << 25));   // pad_amount: 4 DWORDs (16B)
  g1[1] = (int)(64u << 16);                       // tensor_dim0 = 64
  g1[2] = (int)(128u << 16);                      // dim0 hi=0 | tensor_dim1 lo=128
  g1[3] = (int)(64u << 16);                       // dim1 hi=0 | tile_dim0 = 64
  g1[4] = (int)128;                               // tile_dim1 = 128, tile_dim2=0
  g1[5] = (int)row_stride_elems;                  // tensor_dim0_stride[31:0]
  g1[6] = 0;
  g1[7] = 0;
  i32x4 z4 = {0, 0, 0, 0};
#if defined(__clang_major__) && (__clang_major__ >= 23)
  i32x8 z8 = {0, 0, 0, 0, 0, 0, 0, 0};
  __builtin_amdgcn_tensor_load_to_lds(g0, g1, z4, z4, z8, 0);
#else
  __builtin_amdgcn_tensor_load_to_lds(g0, g1, z4, z4, 0);
#endif
}

DEV void tdm_wait0() {
#if __has_builtin(__builtin_amdgcn_s_wait_tensorcnt)
  __builtin_amdgcn_s_wait_tensorcnt(0);
#else
  asm volatile("s_wait_tensorcnt 0x0" ::: "memory");
#endif
}
DEV void tdm_wait2() {  // allow the 2 in-flight loads of the *next* tile
#if __has_builtin(__builtin_amdgcn_s_wait_tensorcnt)
  __builtin_amdgcn_s_wait_tensorcnt(2);
#else
  asm volatile("s_wait_tensorcnt 0x2" ::: "memory");
#endif
}
#endif  // USE_TDM

// ---------------------------------------------------------------------------
// fp32 -> bf16 bulk conversion
// ---------------------------------------------------------------------------
__global__ void cvt_f32_bf16(const float* __restrict__ in,
                             unsigned short* __restrict__ out, int n) {
  int i = blockIdx.x * blockDim.x + threadIdx.x;
  const int stride = gridDim.x * blockDim.x;
  for (; i < n; i += stride) out[i] = f2bf(in[i]);
}

// ---------------------------------------------------------------------------
// C[M,N] = A[M,K] (row-major bf16) x B[N,K]^T (row-major bf16)
// 128x128 output tile per WG; 8 waves; wave = 2(M) x 4(N) WMMA tiles.
// K-step 64 with double-buffered TDM: tile s+1 is always in flight while
// tile s is being consumed; tile s+2 is issued into the vacated buffer.
// mode 0: scatter into Q(scaled)/K as (B,H,N,d) and V^T as (B,H,d,N), bf16.
// mode 1: f32 output + bias (final projection).
// ---------------------------------------------------------------------------
__global__ __launch_bounds__(256)
void gemm_wmma_bf16(const unsigned short* __restrict__ A,
                    const unsigned short* __restrict__ Bm,
                    int K, int mode,
                    unsigned short* __restrict__ Qb,
                    unsigned short* __restrict__ Kb,
                    unsigned short* __restrict__ VTb,
                    float* __restrict__ Out,
                    const float* __restrict__ bias) {
  __shared__ alignas(16) unsigned short As[2][128][80];
  __shared__ alignas(16) unsigned short Bs[2][128][80];

  const int tid = threadIdx.x;
  const int lane = tid & 31;
  const int wv = tid >> 5;
  const int wm = wv & 3;                 // 4 waves along M
  const int wn = wv >> 2;                // 2 waves along N
  const int l15 = lane & 15;
  const int hi = lane >> 4;
  const int mBase = blockIdx.y * 128;
  const int nBase = blockIdx.x * 128;
  const int waveM = wm * 32;
  const int waveN = wn * 64;
  const int STEPS = K >> 6;              // K / 64

  v8f acc[2][4];
  for (int i = 0; i < 2; ++i)
    for (int j = 0; j < 4; ++j) acc[i][j] = vzero8();

#if USE_TDM
  if (wv == 0) {                         // prologue: tiles 0 and 1 in flight
    tdm_load_tile_128x64(A + (size_t)mBase * K,
                         lds_byte_offset(&As[0][0][0]), (unsigned)K);
    tdm_load_tile_128x64(Bm + (size_t)nBase * K,
                         lds_byte_offset(&Bs[0][0][0]), (unsigned)K);
    if (STEPS > 1) {
      tdm_load_tile_128x64(A + (size_t)mBase * K + 64,
                           lds_byte_offset(&As[1][0][0]), (unsigned)K);
      tdm_load_tile_128x64(Bm + (size_t)nBase * K + 64,
                           lds_byte_offset(&Bs[1][0][0]), (unsigned)K);
    }
  }
#endif

  for (int s = 0; s < STEPS; ++s) {
    const int cur = s & 1;
#if USE_TDM
    if (wv == 0) {
      if (s + 1 < STEPS) tdm_wait2();    // tile s done; tile s+1 still in flight
      else               tdm_wait0();    // last tile: drain everything
    }
    __syncthreads();                     // publish buffer `cur` to all waves
#else
    for (int c = tid; c < 1024; c += 256) {
      const int row = c >> 3;
      const int cc = (c & 7) * 8;
      const int lc = cc + ((cc >= 32) ? 8 : 0);   // padded LDS row layout
      *(u32x4*)&As[cur][row][lc] =
          *(const u32x4*)(A + (size_t)(mBase + row) * K + s * 64 + cc);
      *(u32x4*)&Bs[cur][row][lc] =
          *(const u32x4*)(Bm + (size_t)(nBase + row) * K + s * 64 + cc);
    }
    __syncthreads();
#endif

    for (int ks = 0; ks < 2; ++ks) {     // two 16x16x32 sub-steps per stage
      const int cb = ks * 40;            // k=32..63 lives at element 40
      FragBF bf[4];
      for (int nt = 0; nt < 4; ++nt) {
        const int nr = waveN + nt * 16 + l15;
        bf[nt].q[0] = *(const u32x4*)&Bs[cur][nr][cb + hi * 16];
        bf[nt].q[1] = *(const u32x4*)&Bs[cur][nr][cb + hi * 16 + 8];
      }
      for (int mt = 0; mt < 2; ++mt) {
        FragBF af;
        const int mr = waveM + mt * 16 + l15;
        af.q[0] = *(const u32x4*)&As[cur][mr][cb + hi * 16];
        af.q[1] = *(const u32x4*)&As[cur][mr][cb + hi * 16 + 8];
        for (int nt = 0; nt < 4; ++nt)
          acc[mt][nt] = wmma_bf16(af.v, bf[nt].v, acc[mt][nt]);
      }
    }
    __syncthreads();                     // all waves done reading buffer `cur`
#if USE_TDM
    if (wv == 0 && s + 2 < STEPS) {      // refill the vacated buffer
      const int kn = (s + 2) * 64;
      tdm_load_tile_128x64(A + (size_t)mBase * K + kn,
                           lds_byte_offset(&As[cur][0][0]), (unsigned)K);
      tdm_load_tile_128x64(Bm + (size_t)nBase * K + kn,
                           lds_byte_offset(&Bs[cur][0][0]), (unsigned)K);
    }
#endif
  }

  // Epilogue: C/D layout -> lane l15 = column, VGPR r = row r (+8 for hi lanes)
  for (int mt = 0; mt < 2; ++mt) {
    for (int nt = 0; nt < 4; ++nt) {
      const v8f v = acc[mt][nt];
      const int n = nBase + waveN + nt * 16 + l15;
      const int m0 = mBase + waveM + mt * 16 + hi * 8;
      if (mode == 0) {
        const int which = n >> 10;       // 0=q 1=k 2=v
        const int h = (n >> 6) & 15;
        const int dd = n & 63;
        for (int r = 0; r < 8; ++r) {
          const int m = m0 + r;
          const int b = m >> 10;
          const int nn = m & 1023;
          if (which == 0) {
            Qb[((size_t)(b * 16 + h) * 1024 + nn) * 64 + dd] = f2bf(v[r] * 0.125f);
          } else if (which == 1) {
            Kb[((size_t)(b * 16 + h) * 1024 + nn) * 64 + dd] = f2bf(v[r]);
          } else {
            VTb[((size_t)(b * 16 + h) * 64 + dd) * 1024 + nn] = f2bf(v[r]);
          }
        }
      } else {
        const float bb = bias[n];
        for (int r = 0; r < 8; ++r)
          Out[(size_t)(m0 + r) * 1024 + n] = v[r] + bb;
      }
    }
  }
}

// ---------------------------------------------------------------------------
// Flash attention: one (b,h,128-row) slice per block; each wave owns 16 query
// rows; iterate 32 keys at a time with online softmax. Q resident in
// registers, K/V^T fragments loaded directly from global (contiguous per
// lane), P transposed through a 1280B (16x40 bf16) per-wave LDS buffer (in-wave DS order,
// no barriers in the inner loop).
// ---------------------------------------------------------------------------
__global__ __launch_bounds__(256)
void flash_attn_bf16(const unsigned short* __restrict__ Qb,
                     const unsigned short* __restrict__ Kb,
                     const unsigned short* __restrict__ VTb,
                     unsigned short* __restrict__ AOb) {
  __shared__ alignas(16) unsigned short P[8][16][40];
  const int tid = threadIdx.x;
  const int lane = tid & 31;
  const int wv = tid >> 5;
  const int l15 = lane & 15;
  const int hi = lane >> 4;
  const int blk = blockIdx.x;            // [0, B*H*8)
  const int qblk = blk & 7;
  const int bh = blk >> 3;               // b*16 + h
  const int h = bh & 15;
  const int b = bh >> 4;
  const int qbase = qblk * 128 + wv * 16;

  const unsigned short* Qh = Qb + (size_t)bh * 1024 * 64;
  const unsigned short* Kh = Kb + (size_t)bh * 1024 * 64;
  const unsigned short* Vh = VTb + (size_t)bh * 64 * 1024;

  // Q A-fragments for d in [0,32) and [32,64): resident for the whole kernel
  FragBF aq[2];
  {
    const unsigned short* qp = Qh + (size_t)(qbase + l15) * 64 + hi * 16;
    aq[0].q[0] = *(const u32x4*)(qp);
    aq[0].q[1] = *(const u32x4*)(qp + 8);
    aq[1].q[0] = *(const u32x4*)(qp + 32);
    aq[1].q[1] = *(const u32x4*)(qp + 40);
  }

  float mrow[8], lrow[8];
  for (int r = 0; r < 8; ++r) { mrow[r] = -3.0e38f; lrow[r] = 0.f; }
  v8f oacc[4];
  for (int t = 0; t < 4; ++t) oacc[t] = vzero8();

  for (int kb = 0; kb < 1024; kb += 32) {
    v8f s0 = vzero8(), s1 = vzero8();
    {  // S = Q K^T for 32 keys (two 16x16 tiles, K-dim = d = 64 -> 2 steps)
      const unsigned short* kp = Kh + (size_t)(kb + l15) * 64 + hi * 16;
      FragBF bk;
      bk.q[0] = *(const u32x4*)(kp);
      bk.q[1] = *(const u32x4*)(kp + 8);
      s0 = wmma_bf16(aq[0].v, bk.v, s0);
      bk.q[0] = *(const u32x4*)(kp + 32);
      bk.q[1] = *(const u32x4*)(kp + 40);
      s0 = wmma_bf16(aq[1].v, bk.v, s0);
      const unsigned short* kp2 = kp + 16 * 64;
      bk.q[0] = *(const u32x4*)(kp2);
      bk.q[1] = *(const u32x4*)(kp2 + 8);
      s1 = wmma_bf16(aq[0].v, bk.v, s1);
      bk.q[0] = *(const u32x4*)(kp2 + 32);
      bk.q[1] = *(const u32x4*)(kp2 + 40);
      s1 = wmma_bf16(aq[1].v, bk.v, s1);
    }
    // Online softmax across these 32 keys (rows live in VGPR index r)
    for (int r = 0; r < 8; ++r) {
      const float v0 = s0[r], v1 = s1[r];
      const float mx = rmax16(fmaxf(v0, v1));
      const float nm = fmaxf(mrow[r], mx);
      const float corr = __expf(mrow[r] - nm);
      const float p0 = __expf(v0 - nm);
      const float p1 = __expf(v1 - nm);
      lrow[r] = lrow[r] * corr + rsum16(p0 + p1);
      mrow[r] = nm;
      for (int t = 0; t < 4; ++t) oacc[t][r] *= corr;
      const int row = r + hi * 8;
      P[wv][row][l15] = f2bf(p0);          // C-layout -> LDS (row-major 16x32)
      P[wv][row][16 + l15] = f2bf(p1);
    }
    asm volatile("" ::: "memory");  // order P stores before reload (in-wave DS)
    FragBF ap;  // P as A-fragment (16 rows x 32 keys)
    ap.q[0] = *(const u32x4*)&P[wv][l15][hi * 16];
    ap.q[1] = *(const u32x4*)&P[wv][l15][hi * 16 + 8];
    for (int t = 0; t < 4; ++t) {   // O += P V : B-frag from V^T, keys contiguous
      const unsigned short* vp = Vh + (size_t)(t * 16 + l15) * 1024 + kb + hi * 16;
      FragBF bv;
      bv.q[0] = *(const u32x4*)(vp);
      bv.q[1] = *(const u32x4*)(vp + 8);
      oacc[t] = wmma_bf16(ap.v, bv.v, oacc[t]);
    }
  }
  for (int t = 0; t < 4; ++t) {
    for (int r = 0; r < 8; ++r) {
      const int row = qbase + r + hi * 8;
      const int col = h * 64 + t * 16 + l15;
      AOb[(size_t)(b * 1024 + row) * 1024 + col] = f2bf(oacc[t][r] / lrow[r]);
    }
  }
}

// ---------------------------------------------------------------------------
extern "C" void kernel_launch(void* const* d_in, const int* in_sizes, int n_in,
                              void* d_out, int out_size, void* d_ws, size_t ws_size,
                              hipStream_t stream) {
  (void)in_sizes; (void)n_in; (void)out_size; (void)ws_size;
  const float* x      = (const float*)d_in[0];   // (8,1024,1024)
  const float* w_qkv  = (const float*)d_in[1];   // (3072,1024)
  const float* w_proj = (const float*)d_in[2];   // (1024,1024)
  const float* b_proj = (const float*)d_in[3];   // (1024,)
  float* out = (float*)d_out;

  char* w = (char*)d_ws;
  const size_t MB = 1ull << 20;
  unsigned short* xb     = (unsigned short*)(w + 0 * MB);   // 16 MB
  unsigned short* wqkvb  = (unsigned short*)(w + 16 * MB);  //  6 MB
  unsigned short* wprojb = (unsigned short*)(w + 22 * MB);  //  2 MB
  unsigned short* Qb     = (unsigned short*)(w + 24 * MB);  // 16 MB (B,H,N,d)
  unsigned short* Kb     = (unsigned short*)(w + 40 * MB);  // 16 MB (B,H,N,d)
  unsigned short* VTb    = (unsigned short*)(w + 56 * MB);  // 16 MB (B,H,d,N)
  unsigned short* AOb    = (unsigned short*)(w + 72 * MB);  // 16 MB (B,N,C)

  cvt_f32_bf16<<<4096, 256, 0, stream>>>(x, xb, 8 * 1024 * 1024);
  cvt_f32_bf16<<<4096, 256, 0, stream>>>(w_qkv, wqkvb, 3072 * 1024);
  cvt_f32_bf16<<<2048, 256, 0, stream>>>(w_proj, wprojb, 1024 * 1024);

  dim3 g_qkv(3072 / 128, 8192 / 128);   // (24, 64)
  gemm_wmma_bf16<<<g_qkv, 256, 0, stream>>>(xb, wqkvb, 1024, 0,
                                            Qb, Kb, VTb, nullptr, nullptr);

  flash_attn_bf16<<<8 * 16 * 8, 256, 0, stream>>>(Qb, Kb, VTb, AOb);

  dim3 g_proj(1024 / 128, 8192 / 128);  // (8, 64)
  gemm_wmma_bf16<<<g_proj, 256, 0, stream>>>(AOb, wprojb, 1024, 1,
                                             nullptr, nullptr, nullptr, out, b_proj);
}
